// HeteroLayer_33578054320522
// MI455X (gfx1250) — matrix operand order repaired and checked by reading, not compiled, not run on hardware
//
#include <hip/hip_runtime.h>
#include <hip/hip_bf16.h>
#include <math.h>

// ---------------------------------------------------------------------------
// 2-layer GCN for MI455X (gfx1250, wave32, WMMA).
//   layer: support = X @ W  (bf16 WMMA, f32 accum)
//          agg     = scatter_add(support[src] * ew, dst)   (f32 atomics)
//          h       = elu(agg + b)
// ---------------------------------------------------------------------------

#define N_NODES 100000
#define N_EDGES 3200000
#define NF      256                      // NFEAT == NHID == 256

typedef __attribute__((ext_vector_type(16))) __bf16 v16bf;
typedef __attribute__((ext_vector_type(8)))  float  v8f;

union BFrag { v16bf bf; unsigned int u[8]; };

// f32 -> bf16: rely on hardware V_CVT_PK_BF16_F32 via the native cast.
static __device__ __forceinline__ __bf16 f2bf(float f) {
    return (__bf16)f;
}

// ---------------------------------------------------------------------------
// Convert + transpose weights: Wt[n][k] = bf16(W[k][n]).  256x256, one shot.
// ---------------------------------------------------------------------------
__global__ void wconv_kernel(const float* __restrict__ W, __bf16* __restrict__ Wt) {
    int n = blockIdx.x;      // 256 blocks
    int k = threadIdx.x;     // 256 threads
    Wt[n * NF + k] = f2bf(W[k * NF + n]);
}

// ---------------------------------------------------------------------------
// GEMM: C[M x 256] = A[M x 256] (f32, converted inline to bf16) @ Wt^T
// Wt is stored [N][K] bf16 so B-fragments are contiguous 32B loads.
// Block: 256 threads = 8 waves. Block tile: 32(M) x 256(N).
// Wave w: m0 = 32*bx + 16*(w&1), n-strip = 64*(w>>1); 4 N-tiles per wave.
// Grid: M/32 blocks (M = 100000 -> 3125).
// Inner loop: batch all 4 B-fragment loads (8x b128 in one clause), then
// 4 back-to-back WMMAs so loads/VALU co-execute with the XDL pipe.
// ---------------------------------------------------------------------------
__global__ void gemm_bf16_wmma_kernel(const float* __restrict__ A,
                                      const __bf16* __restrict__ Wt,
                                      float* __restrict__ C) {
    const int lane  = threadIdx.x & 31;
    const int w     = threadIdx.x >> 5;
    const int m0    = blockIdx.x * 32 + (w & 1) * 16;
    const int nbase = (w >> 1) * 64;

    const int halfsel = lane >> 4;        // 0: lanes 0-15, 1: lanes 16-31
    const int l15     = lane & 15;

    v8f acc[4];
    #pragma unroll
    for (int t = 0; t < 4; ++t)
        #pragma unroll
        for (int r = 0; r < 8; ++r) acc[t][r] = 0.0f;

    const int arow = m0 + l15;            // A row owned by this lane
    const int akb  = halfsel * 16;        // K sub-base for A fragment

    const float*  abase = A + (size_t)arow * NF + akb;
    const __bf16* bbase = Wt + (size_t)(nbase + l15) * NF + halfsel * 16;

    for (int ks = 0; ks < NF; ks += 32) {
        // ---- A fragment (16x32 bf16), contiguous k = 16*halfsel + s, same as B ----
        const float* ap = abase + ks;
        float4 a0 = *(const float4*)(ap + 0);
        float4 a1 = *(const float4*)(ap + 4);
        float4 a2 = *(const float4*)(ap + 8);
        float4 a3 = *(const float4*)(ap + 12);

        // ---- all 4 B fragments first: 8x global_load_b128 in one clause ----
        BFrag bf[4];
        #pragma unroll
        for (int t = 0; t < 4; ++t) {
            // N-tile t: rows (nbase + 16t + l15) of Wt, 16 contiguous bf16
            const uint4* q = (const uint4*)(bbase + (size_t)(16 * t) * NF + ks);
            uint4 b0 = q[0];
            uint4 b1 = q[1];
            bf[t].u[0] = b0.x; bf[t].u[1] = b0.y; bf[t].u[2] = b0.z; bf[t].u[3] = b0.w;
            bf[t].u[4] = b1.x; bf[t].u[5] = b1.y; bf[t].u[6] = b1.z; bf[t].u[7] = b1.w;
        }

        v16bf afrag;
        afrag[0]  = f2bf(a0.x); afrag[1]  = f2bf(a0.y);
        afrag[2]  = f2bf(a0.z); afrag[3]  = f2bf(a0.w);
        afrag[4]  = f2bf(a1.x); afrag[5]  = f2bf(a1.y);
        afrag[6]  = f2bf(a1.z); afrag[7]  = f2bf(a1.w);
        afrag[8]  = f2bf(a2.x); afrag[9]  = f2bf(a2.y);
        afrag[10] = f2bf(a2.z); afrag[11] = f2bf(a2.w);
        afrag[12] = f2bf(a3.x); afrag[13] = f2bf(a3.y);
        afrag[14] = f2bf(a3.z); afrag[15] = f2bf(a3.w);

        #pragma unroll
        for (int t = 0; t < 4; ++t) {
            acc[t] = __builtin_amdgcn_wmma_f32_16x16x32_bf16(
                /*neg_a=*/false, afrag, /*neg_b=*/false, bf[t].bf,
                /*c_mod=*/(short)0, acc[t], /*reuse_a=*/false, /*reuse_b=*/false);
        }
    }

    // ---- store C (f32 C/D layout: VGPR r -> M=r+8*halfsel, N=lane&15) ----
    const int crow = m0 + halfsel * 8;
    #pragma unroll
    for (int t = 0; t < 4; ++t) {
        const int cn = nbase + 16 * t + l15;
        float* cp = C + (size_t)crow * NF + cn;
        #pragma unroll
        for (int r = 0; r < 8; ++r)
            cp[(size_t)r * NF] = acc[t][r];
    }
}

// ---------------------------------------------------------------------------
// Zero-fill (float4), exact sizing: count4 = n/4 float4 elements.
// ---------------------------------------------------------------------------
__global__ void zero_kernel(float4* __restrict__ p) {
    size_t i = (size_t)blockIdx.x * blockDim.x + threadIdx.x;
    p[i] = make_float4(0.f, 0.f, 0.f, 0.f);
}

// ---------------------------------------------------------------------------
// SpMM scatter: one wave32 per edge. Lane l owns features [8l, 8l+8).
// Gather 1KB row (2x float4 per lane), scale, 8x global_atomic_add_f32.
// ---------------------------------------------------------------------------
__global__ void spmm_scatter_kernel(const float* __restrict__ sup,
                                    const float* __restrict__ ew,
                                    const int*   __restrict__ esrc,
                                    const int*   __restrict__ edst,
                                    float* __restrict__ agg) {
    const int e    = (blockIdx.x * blockDim.x + threadIdx.x) >> 5; // edge id
    const int lane = threadIdx.x & 31;
    if (e >= N_EDGES) return;

    const int   s = esrc[e];
    const int   d = edst[e];
    const float w = ew[e];

    const float4* sp = (const float4*)(sup + (size_t)s * NF + lane * 8);
    float4 a = sp[0];
    float4 b = sp[1];

    float* dp = agg + (size_t)d * NF + lane * 8;
    atomicAdd(dp + 0, a.x * w);
    atomicAdd(dp + 1, a.y * w);
    atomicAdd(dp + 2, a.z * w);
    atomicAdd(dp + 3, a.w * w);
    atomicAdd(dp + 4, b.x * w);
    atomicAdd(dp + 5, b.y * w);
    atomicAdd(dp + 6, b.z * w);
    atomicAdd(dp + 7, b.w * w);
}

// ---------------------------------------------------------------------------
// out[i] = elu(agg[i] + b[i % 256]), float4 per thread, in-place safe.
// ---------------------------------------------------------------------------
__global__ void bias_elu_kernel(const float* __restrict__ agg,
                                const float* __restrict__ bias,
                                float* __restrict__ out) {
    size_t i4   = (size_t)blockIdx.x * blockDim.x + threadIdx.x;
    size_t base = i4 * 4;
    int    col  = (int)(base & (NF - 1));

    float4 v  = *(const float4*)(agg + base);
    float4 bb = *(const float4*)(bias + col);
    float r0 = v.x + bb.x, r1 = v.y + bb.y, r2 = v.z + bb.z, r3 = v.w + bb.w;
    r0 = r0 > 0.f ? r0 : (expf(r0) - 1.f);
    r1 = r1 > 0.f ? r1 : (expf(r1) - 1.f);
    r2 = r2 > 0.f ? r2 : (expf(r2) - 1.f);
    r3 = r3 > 0.f ? r3 : (expf(r3) - 1.f);
    *(float4*)(out + base) = make_float4(r0, r1, r2, r3);
}

// ---------------------------------------------------------------------------
// Host-side orchestration (graph-capture safe: kernels only, all on stream).
// ---------------------------------------------------------------------------
extern "C" void kernel_launch(void* const* d_in, const int* in_sizes, int n_in,
                              void* d_out, int out_size, void* d_ws, size_t ws_size,
                              hipStream_t stream) {
    const float* x  = (const float*)d_in[0];
    const float* W1 = (const float*)d_in[1];
    const float* b1 = (const float*)d_in[2];
    const float* W2 = (const float*)d_in[3];
    const float* b2 = (const float*)d_in[4];
    const float* ew = (const float*)d_in[5];
    const int* esrc = (const int*)d_in[6];
    const int* edst = (const int*)d_in[7];
    float* out = (float*)d_out;

    const size_t SUP_BYTES = (size_t)N_NODES * NF * sizeof(float);   // 102.4 MB
    const size_t WT_BYTES  = (size_t)NF * NF * sizeof(__bf16);       // 128 KB

    char* ws = (char*)d_ws;
    float*  support = (float*)ws;                                    // gather source
    float*  agg     = (float*)(ws + SUP_BYTES);                      // layer-1 agg / h1
    __bf16* W1t     = (__bf16*)(ws + 2 * SUP_BYTES);
    __bf16* W2t     = (__bf16*)(ws + 2 * SUP_BYTES + WT_BYTES);

    const int GEMM_BLOCKS  = N_NODES / 32;             // 3125
    const int ZERO_BLOCKS  = (N_NODES * NF / 4) / 256; // 25000
    const int EW_BLOCKS    = N_EDGES / 8;              // 400000 (8 waves/block)
    const int ELU_BLOCKS   = ZERO_BLOCKS;              // 25000

    // weight prep (bf16, transposed for contiguous B fragments)
    wconv_kernel<<<NF, NF, 0, stream>>>(W1, W1t);
    wconv_kernel<<<NF, NF, 0, stream>>>(W2, W2t);

    // ---- layer 1 ----
    gemm_bf16_wmma_kernel<<<GEMM_BLOCKS, 256, 0, stream>>>(x, W1t, support);
    zero_kernel<<<ZERO_BLOCKS, 256, 0, stream>>>((float4*)agg);
    spmm_scatter_kernel<<<EW_BLOCKS, 256, 0, stream>>>(support, ew, esrc, edst, agg);
    bias_elu_kernel<<<ELU_BLOCKS, 256, 0, stream>>>(agg, b1, agg);   // h1 in place

    // ---- layer 2 ----
    gemm_bf16_wmma_kernel<<<GEMM_BLOCKS, 256, 0, stream>>>(agg, W2t, support);
    zero_kernel<<<ZERO_BLOCKS, 256, 0, stream>>>((float4*)out);
    spmm_scatter_kernel<<<EW_BLOCKS, 256, 0, stream>>>(support, ew, esrc, edst, out);
    bias_elu_kernel<<<ELU_BLOCKS, 256, 0, stream>>>(out, b2, out);
}
